// BlockV2_49830210568839
// MI455X (gfx1250) — compile-verified, matrix operand order checked by reading, not run on hardware
//
#include <hip/hip_runtime.h>
#include <math.h>

// ---------------------------------------------------------------------------
// Types for CDNA5 WMMA (wave32, 16x16x32 bf16 -> f32 accumulate)
// ---------------------------------------------------------------------------
typedef __attribute__((ext_vector_type(16))) __bf16 v16bf;
typedef __attribute__((ext_vector_type(8)))  float  v8f;

struct alignas(16) U4 { unsigned x, y, z, w; };
struct alignas(8)  U2 { unsigned x, y; };
union FragU { U4 q[2]; v16bf v; };

__device__ __forceinline__ unsigned short f2bf(float f) {
  unsigned u = __float_as_uint(f);
  u += 0x7FFFu + ((u >> 16) & 1u);      // round-to-nearest-even
  return (unsigned short)(u >> 16);
}
__device__ __forceinline__ unsigned pk2(float a, float b) {
  return (unsigned)f2bf(a) | ((unsigned)f2bf(b) << 16);
}

// CDNA5 async global->LDS 16B copy (ASYNCcnt-tracked); inline asm per bridge doc
__device__ __forceinline__ void async_b128(unsigned lds_off, unsigned long long gaddr) {
  asm volatile("global_load_async_to_lds_b128 %0, %1, off"
               :: "v"(lds_off), "v"(gaddr) : "memory");
}
__device__ __forceinline__ void wait_async_le3() {
  asm volatile("s_wait_asynccnt 0x3" ::: "memory");
}
__device__ __forceinline__ void wait_async_0() {
  asm volatile("s_wait_asynccnt 0x0" ::: "memory");
}

// ---------------------------------------------------------------------------
// GEMM: C = act(A[M,K]bf16 @ W[N,K]bf16^T + bias), C written as f32 and/or bf16
// Block tile 128x64, K-step 32, 8 waves (4x2), each wave 32x32 (2x2 wmma tiles)
// Tiles filled by GLOBAL_LOAD_ASYNC_TO_LDS_B128, double-buffered in LDS.
// ---------------------------------------------------------------------------
template<bool RELU>
__global__ __launch_bounds__(256) void gemm_bf16_wmma(
    const unsigned short* __restrict__ A, const unsigned short* __restrict__ W,
    const float* __restrict__ bias, float* __restrict__ Cf,
    unsigned short* __restrict__ Cb, int M, int N, int K)
{
  constexpr int BM = 128, BN = 64, BK = 32;
  constexpr int LDTB = 80;                 // LDS row pitch bytes (64B data + 16B pad)
  constexpr int ABYTES = BM * LDTB;        // 10240
  constexpr int BBYTES = BN * LDTB;        // 5120
  constexpr int BUFB   = ABYTES + BBYTES;  // 15360 per buffer
  __shared__ unsigned short lds[2 * BUFB / 2];

  const unsigned ldsBase = (unsigned)(size_t)(void*)lds;  // low 32b = LDS offset
  const int tid  = threadIdx.x;
  const int bm   = blockIdx.x * BM;
  const int bn   = blockIdx.y * BN;
  const int wave = tid >> 5;
  const int lane = tid & 31;
  const int wm   = (wave >> 1) * 32;   // 0,32,64,96
  const int wn   = (wave & 1) * 32;    // 0,32

  // async-copy descriptors (constant over K; per-iter just add k0*2 bytes)
  const int crow = tid >> 2, ccol = tid & 3;   // 16B chunk: row, col-of-4
  const unsigned aoff0 = ldsBase + crow * LDTB + ccol * 16;
  const unsigned aoff1 = aoff0 + 64 * LDTB;
  const unsigned boff  = ldsBase + ABYTES + crow * LDTB + ccol * 16;
  const unsigned long long ag0 =
      (unsigned long long)(size_t)A + ((size_t)(bm + crow) * K + ccol * 8) * 2;
  const unsigned long long ag1 =
      (unsigned long long)(size_t)A + ((size_t)(bm + 64 + crow) * K + ccol * 8) * 2;
  const unsigned long long bg =
      (unsigned long long)(size_t)W + ((size_t)(bn + crow) * K + ccol * 8) * 2;

  v8f acc[2][2];
  const v8f vz = {0.f,0.f,0.f,0.f,0.f,0.f,0.f,0.f};
  acc[0][0]=vz; acc[0][1]=vz; acc[1][0]=vz; acc[1][1]=vz;

  // fragment addressing per ISA 7.12.2: lane&15 = M/N row, K-octet split
  const int koff = (lane >= 16) ? 8 : 0;
  const int mrow = lane & 15;

  const int NS = K / BK;
  // prologue: tile 0 into buffer 0
  {
    async_b128(aoff0, ag0);
    async_b128(aoff1, ag1);
    async_b128(boff,  bg);
  }
  for (int s = 0; s < NS; ++s) {
    const int cur = s & 1;
    if (s + 1 < NS) {
      const unsigned bb = (unsigned)((cur ^ 1) * BUFB);
      const size_t kb = (size_t)(s + 1) * BK * 2;
      async_b128(aoff0 + bb, ag0 + kb);
      async_b128(aoff1 + bb, ag1 + kb);
      async_b128(boff  + bb, bg  + kb);
      wait_async_le3();                 // in-order: tile s is now resident
    } else {
      wait_async_0();
    }
    __syncthreads();

    const unsigned short* As = lds + (size_t)cur * (BUFB / 2);
    const unsigned short* Bs = As + ABYTES / 2;
    FragU af[2], bf[2];
#pragma unroll
    for (int i = 0; i < 2; ++i) {
      const unsigned short* pa = &As[(wm + i*16 + mrow) * (LDTB/2)];
      af[i].q[0] = *(const U4*)(pa + koff);        // K = koff..koff+7
      af[i].q[1] = *(const U4*)(pa + koff + 16);   // K = koff+16..koff+23
      const unsigned short* pb = &Bs[(wn + i*16 + mrow) * (LDTB/2)];
      bf[i].q[0] = *(const U4*)(pb + koff);
      bf[i].q[1] = *(const U4*)(pb + koff + 16);
    }
#pragma unroll
    for (int i = 0; i < 2; ++i)
#pragma unroll
      for (int j = 0; j < 2; ++j)
        acc[i][j] = __builtin_amdgcn_wmma_f32_16x16x32_bf16(
            false, af[i].v, false, bf[j].v, (short)0, acc[i][j], false, false);
    __syncthreads();                    // tile cur free for re-fill next iter
  }

  const int nlane = lane & 15;
  const int mofs  = (lane >= 16) ? 8 : 0;
#pragma unroll
  for (int j = 0; j < 2; ++j) {
    const int n = bn + wn + j * 16 + nlane;
    const float bv = bias ? bias[n] : 0.0f;
#pragma unroll
    for (int i = 0; i < 2; ++i) {
      const int mb = bm + wm + i * 16 + mofs;
#pragma unroll
      for (int r = 0; r < 8; ++r) {
        float v = acc[i][j][r] + bv;
        if (RELU) v = fmaxf(v, 0.0f);
        const size_t idx = (size_t)(mb + r) * N + n;
        if (Cf) Cf[idx] = v;
        if (Cb) Cb[idx] = f2bf(v);
      }
    }
  }
}

// ---------------------------------------------------------------------------
// fp32 -> bf16 pack (weights prepack), float4-granular
// ---------------------------------------------------------------------------
__global__ __launch_bounds__(256) void cvt_bf16_kernel(
    const float* __restrict__ x, unsigned short* __restrict__ y, long long n4)
{
  const size_t idx = (size_t)blockIdx.x * blockDim.x + threadIdx.x;
  if (idx >= (size_t)n4) return;
  float4 v = ((const float4*)x)[idx];
  U2 p; p.x = pk2(v.x, v.y); p.y = pk2(v.z, v.w);
  ((U2*)y)[idx] = p;
}

// ---------------------------------------------------------------------------
// LayerNorm: one wave32 per row, writes fp32 + bf16 copies
// ---------------------------------------------------------------------------
__global__ __launch_bounds__(256) void ln_kernel(
    const float* __restrict__ x, const float* __restrict__ g,
    const float* __restrict__ b, float* __restrict__ y,
    unsigned short* __restrict__ ybf, int M, int D)
{
  const int wave = threadIdx.x >> 5, lane = threadIdx.x & 31;
  const int row  = blockIdx.x * 8 + wave;
  if (row >= M) return;
  const float4* xr = (const float4*)(x + (size_t)row * D);
  float4 v[4];
  float s = 0.f, ss = 0.f;
#pragma unroll
  for (int j = 0; j < 4; ++j) {
    v[j] = xr[lane + 32 * j];
    s  += v[j].x + v[j].y + v[j].z + v[j].w;
    ss += v[j].x*v[j].x + v[j].y*v[j].y + v[j].z*v[j].z + v[j].w*v[j].w;
  }
#pragma unroll
  for (int o = 16; o > 0; o >>= 1) {
    s  += __shfl_xor(s,  o, 32);
    ss += __shfl_xor(ss, o, 32);
  }
  const float invD = 1.0f / (float)D;
  const float mu   = s * invD;
  const float var  = ss * invD - mu * mu;
  const float rstd = rsqrtf(var + 1e-5f);
  float4* yr = (float4*)(y + (size_t)row * D);
  U2* yb = (U2*)(ybf + (size_t)row * D);
  const float4* g4 = (const float4*)g;
  const float4* b4 = (const float4*)b;
#pragma unroll
  for (int j = 0; j < 4; ++j) {
    const int c = lane + 32 * j;
    float4 gg = g4[c], bb = b4[c], o4;
    o4.x = (v[j].x - mu) * rstd * gg.x + bb.x;
    o4.y = (v[j].y - mu) * rstd * gg.y + bb.y;
    o4.z = (v[j].z - mu) * rstd * gg.z + bb.z;
    o4.w = (v[j].w - mu) * rstd * gg.w + bb.w;
    yr[c] = o4;
    U2 p; p.x = pk2(o4.x, o4.y); p.y = pk2(o4.z, o4.w);
    yb[c] = p;
  }
}

// ---------------------------------------------------------------------------
// Depthwise causal conv (K=4) + bias; output bf16 (consumed only by GEMM)
// ---------------------------------------------------------------------------
__global__ __launch_bounds__(256) void dwconv_kernel(
    const float* __restrict__ x, const float* __restrict__ w,
    const float* __restrict__ bias, unsigned short* __restrict__ y,
    int B, int T, int D)
{
  const int D4 = D >> 2;
  const size_t idx = (size_t)blockIdx.x * blockDim.x + threadIdx.x;
  const size_t total = (size_t)B * T * D4;
  if (idx >= total) return;
  const int d4 = (int)(idx % D4);
  const size_t bt = idx / D4;
  const int t = (int)(bt % T);
  const int b = (int)(bt / T);
  const float4* x4 = (const float4*)x;
  const float4* w4 = (const float4*)w;     // [4][D4]
  float4 acc = ((const float4*)bias)[d4];
#pragma unroll
  for (int j = 0; j < 4; ++j) {
    const int tt = t - 3 + j;
    if (tt >= 0) {
      float4 xv = x4[((size_t)b * T + tt) * D4 + d4];
      float4 wv = w4[j * D4 + d4];
      acc.x += xv.x * wv.x; acc.y += xv.y * wv.y;
      acc.z += xv.z * wv.z; acc.w += xv.w * wv.w;
    }
  }
  U2 p; p.x = pk2(acc.x, acc.y); p.y = pk2(acc.z, acc.w);
  ((U2*)y)[idx] = p;
}

// ---------------------------------------------------------------------------
// minGRU log-space scan, 3-phase chunked (CHUNK=64, NCH=T/64)
// log_h_t = logaddexp(log_c_t + log_h_{t-1}, log_z_t + log_g(h_x_t)), h0=0.5
// ---------------------------------------------------------------------------
#define NEG_BIG (-3.0e38f)

__device__ __forceinline__ float softplusf_(float x) {
  return fmaxf(x, 0.f) + log1pf(__expf(-fabsf(x)));
}
__device__ __forceinline__ float logaddexpf_(float a, float b) {
  float m = fmaxf(a, b);
  return m + log1pf(__expf(fminf(a, b) - m));
}
__device__ __forceinline__ float log_gf(float x) {
  return (x >= 0.f) ? __logf(x + 0.5f) : -softplusf_(-x);
}

__global__ __launch_bounds__(256) void scan_phase1(
    const float* __restrict__ kh, float* __restrict__ sumC,
    float* __restrict__ sumV, int B, int T, int D, int nch, int chunk)
{
  const size_t idx = (size_t)blockIdx.x * blockDim.x + threadIdx.x;
  const size_t total = (size_t)B * D * nch;
  if (idx >= total) return;
  const int d = (int)(idx % D);
  const int c = (int)((idx / D) % nch);
  const int b = (int)(idx / ((size_t)D * nch));
  const size_t base = ((size_t)b * T + (size_t)c * chunk) * (size_t)(2 * D);
  float cc = 0.f, lv = NEG_BIG;
  for (int t = 0; t < chunk; ++t) {
    const float k  = kh[base + (size_t)t * 2 * D + d];
    const float hx = kh[base + (size_t)t * 2 * D + D + d];
    const float lc  = -softplusf_(k);
    const float val = -softplusf_(-k) + log_gf(hx);
    cc += lc;
    lv = logaddexpf_(lc + lv, val);
  }
  const size_t o = ((size_t)b * nch + c) * D + d;
  sumC[o] = cc; sumV[o] = lv;
}

__global__ __launch_bounds__(256) void scan_phase2(
    const float* __restrict__ sumC, const float* __restrict__ sumV,
    float* __restrict__ h0, int B, int D, int nch)
{
  const size_t idx = (size_t)blockIdx.x * blockDim.x + threadIdx.x;
  if (idx >= (size_t)B * D) return;
  const int d = (int)(idx % D);
  const int b = (int)(idx / D);
  float carry = -0.6931471805599453f;   // log(0.5)
  for (int c = 0; c < nch; ++c) {
    const size_t o = ((size_t)b * nch + c) * D + d;
    h0[o] = carry;
    carry = logaddexpf_(sumC[o] + carry, sumV[o]);
  }
}

__global__ __launch_bounds__(256) void scan_phase3(
    const float* __restrict__ kh, const float* __restrict__ h0,
    const float* res, float* out, int B, int T, int D, int nch, int chunk)
{
  const size_t idx = (size_t)blockIdx.x * blockDim.x + threadIdx.x;
  const size_t total = (size_t)B * D * nch;
  if (idx >= total) return;
  const int d = (int)(idx % D);
  const int c = (int)((idx / D) % nch);
  const int b = (int)(idx / ((size_t)D * nch));
  float lh = h0[((size_t)b * nch + c) * D + d];
  const size_t base  = ((size_t)b * T + (size_t)c * chunk) * (size_t)(2 * D);
  const size_t obase = ((size_t)b * T + (size_t)c * chunk) * (size_t)D;
  for (int t = 0; t < chunk; ++t) {
    const float k  = kh[base + (size_t)t * 2 * D + d];
    const float hx = kh[base + (size_t)t * 2 * D + D + d];
    const float lc  = -softplusf_(k);
    const float val = -softplusf_(-k) + log_gf(hx);
    lh = logaddexpf_(lc + lh, val);
    const size_t oo = obase + (size_t)t * D + d;
    out[oo] = __expf(lh) + res[oo];   // same-element RMW: safe when res==out
  }
}

// ---------------------------------------------------------------------------
// Orchestration
// ---------------------------------------------------------------------------
extern "C" void kernel_launch(void* const* d_in, const int* in_sizes, int n_in,
                              void* d_out, int out_size, void* d_ws, size_t ws_size,
                              hipStream_t stream) {
  (void)in_sizes; (void)n_in; (void)out_size; (void)ws_size;
  const int B = 8, T = 4096, D = 512, H = 2048, L = 4;
  const int M = B * T;
  const int NCH = 64, CHUNK = 64;

  const float* x_in  = (const float*)d_in[0];
  const float* dw_w  = (const float*)d_in[1];   // [L][4][D]
  const float* dw_b  = (const float*)d_in[2];   // [L][D]
  const float* pw_w  = (const float*)d_in[3];   // [L][D][D]
  const float* pw_b  = (const float*)d_in[4];   // [L][D]
  const float* ln1_g = (const float*)d_in[5];
  const float* ln1_b = (const float*)d_in[6];
  const float* f_w   = (const float*)d_in[7];   // [L][2D][D]
  const float* ln2_g = (const float*)d_in[8];   // [L][D]
  const float* ln2_b = (const float*)d_in[9];
  const float* w1    = (const float*)d_in[10];  // [L][H][D]
  const float* b1    = (const float*)d_in[11];  // [L][H]
  const float* w2    = (const float*)d_in[12];  // [L][D][H]
  const float* b2    = (const float*)d_in[13];  // [L][D]
  float* out = (float*)d_out;

  const size_t MD = (size_t)M * D;
  const size_t SC = (size_t)B * D * NCH;
  float* ws  = (float*)d_ws;
  float* bx  = ws;            // current activations / residual (f32)
  float* by  = bx + MD;       // LN f32 output (prologue residual)
  float* bt_ = by + MD;       // MLP f32 output (conv input)
  float* kh  = bt_ + MD;      // 2*MD, minGRU gate pre-activations
  float* sc  = kh + 2 * MD;
  float* sv  = sc + SC;
  float* h0  = sv + SC;
  unsigned short* ub       = (unsigned short*)(h0 + SC);
  unsigned short* abf_conv = ub;                 // MD  (conv out, bf16)
  unsigned short* abf_x    = abf_conv + MD;      // MD  (pw-gemm out, bf16)
  unsigned short* abf_ln   = abf_x + MD;         // MD  (LN out, bf16)
  unsigned short* abf_h    = abf_ln + MD;        // 8MD (relu hidden, bf16-only)
  unsigned short* wb_pw    = abf_h + 8 * MD;                       // L*D*D
  unsigned short* wb_f     = wb_pw + (size_t)L * D * D;            // L*2D*D
  unsigned short* wb_w1    = wb_f  + (size_t)L * 2 * D * D;        // L*H*D
  unsigned short* wb_w2    = wb_w1 + (size_t)L * H * D;            // L*D*H

  const dim3 blk(256);
  auto cvt = [&](const float* src, unsigned short* dst, size_t n) {
    const size_t n4 = n / 4;
    hipLaunchKernelGGL(cvt_bf16_kernel, dim3((unsigned)((n4 + 255) / 256)), blk, 0, stream,
                       src, dst, (long long)n4);
  };
  auto gemm = [&](const unsigned short* A, const unsigned short* Wm, const float* bias,
                  float* Cf, unsigned short* Cb, int N, int K, bool relu) {
    dim3 grid(M / 128, N / 64);
    if (relu) hipLaunchKernelGGL((gemm_bf16_wmma<true>),  grid, blk, 0, stream, A, Wm, bias, Cf, Cb, M, N, K);
    else      hipLaunchKernelGGL((gemm_bf16_wmma<false>), grid, blk, 0, stream, A, Wm, bias, Cf, Cb, M, N, K);
  };
  auto lnk = [&](const float* xp, const float* gp, const float* bp,
                 float* yp, unsigned short* ybf) {
    hipLaunchKernelGGL(ln_kernel, dim3(M / 8), blk, 0, stream, xp, gp, bp, yp, ybf, M, D);
  };
  auto conv = [&](const float* xp, int layer, unsigned short* yp) {
    const size_t tot = (size_t)M * (D / 4);
    hipLaunchKernelGGL(dwconv_kernel, dim3((unsigned)((tot + 255) / 256)), blk, 0, stream,
                       xp, dw_w + (size_t)layer * 4 * D, dw_b + (size_t)layer * D, yp, B, T, D);
  };
  auto scan = [&](const float* khp, const float* res, float* op) {
    const size_t t1 = (size_t)B * D * NCH;
    hipLaunchKernelGGL(scan_phase1, dim3((unsigned)((t1 + 255) / 256)), blk, 0, stream,
                       khp, sc, sv, B, T, D, NCH, CHUNK);
    const size_t t2 = (size_t)B * D;
    hipLaunchKernelGGL(scan_phase2, dim3((unsigned)((t2 + 255) / 256)), blk, 0, stream,
                       sc, sv, h0, B, D, NCH);
    hipLaunchKernelGGL(scan_phase3, dim3((unsigned)((t1 + 255) / 256)), blk, 0, stream,
                       khp, h0, res, op, B, T, D, NCH, CHUNK);
  };

  // ---- weight prepack: fp32 -> bf16, once per launch ----
  cvt(pw_w, wb_pw, (size_t)L * D * D);
  cvt(f_w,  wb_f,  (size_t)L * 2 * D * D);
  cvt(w1,   wb_w1, (size_t)L * H * D);
  cvt(w2,   wb_w2, (size_t)L * D * H);

  // ---- layer 0 prologue ----
  conv(x_in, 0, abf_conv);                                     // depthwise conv (bf16 out)
  gemm(abf_conv, wb_pw, pw_b, bx, abf_x, D, D, false);         // pointwise (f32 read by ln1, bf16 unused later - keep f32 path)
  lnk(bx, ln1_g, ln1_b, by, abf_ln);                           // ln1 (f32 residual + bf16)
  gemm(abf_ln, wb_f, nullptr, kh, nullptr, 2 * D, D, false);   // kh = y @ f_w^T
  scan(kh, by, bx);                                            // x = minGRU + residual(LN out)

  // ---- middle layers ----
  for (int i = 0; i < 3; ++i) {
    lnk(bx, ln2_g + (size_t)i * D, ln2_b + (size_t)i * D, by, abf_ln);
    gemm(abf_ln, wb_w1 + (size_t)i * H * D, b1 + (size_t)i * H, nullptr, abf_h, H, D, true);
    gemm(abf_h, wb_w2 + (size_t)i * D * H, b2 + (size_t)i * D, bt_, nullptr, D, H, false);
    conv(bt_, i + 1, abf_conv);
    gemm(abf_conv, wb_pw + (size_t)(i + 1) * D * D, pw_b + (size_t)(i + 1) * D,
         bx, abf_x, D, D, false);
    gemm(abf_x, wb_f + (size_t)(i + 1) * 2 * D * D, nullptr, kh, nullptr, 2 * D, D, false);
    scan(kh, bx, bx);                                          // x = minGRU + residual(conv out)
  }

  // ---- epilogue ----
  lnk(bx, ln2_g + (size_t)3 * D, ln2_b + (size_t)3 * D, by, abf_ln);
  gemm(abf_ln, wb_w1 + (size_t)3 * H * D, b1 + (size_t)3 * H, nullptr, abf_h, H, D, true);
  gemm(abf_h, wb_w2 + (size_t)3 * D * H, b2 + (size_t)3 * D, out, nullptr, D, H, false);
}
